// SelfAttention_22926535426238
// MI455X (gfx1250) — matrix operand order repaired and checked by reading, not compiled, not run on hardware
//
#include <hip/hip_runtime.h>

// ---------------------------------------------------------------------------
// MI455X (gfx1250) self-attention: bf16 WMMA GEMMs with TDM (tensor_load_to_lds)
// double-buffered LDS staging + register-resident flash attention (wave32).
// ---------------------------------------------------------------------------

#define B_   2
#define T_   2048
#define D_   2048
#define NQ_  16
#define NK_  4
#define H_   128
#define BT_  (B_ * T_)                       // 4096
#define CQKV_ (NQ_ * H_ + 2 * NK_ * H_)      // 3072

typedef __bf16 bf16;
typedef __attribute__((ext_vector_type(16))) __bf16 v16bf;
typedef __attribute__((ext_vector_type(8)))  float  v8f;
typedef unsigned __attribute__((ext_vector_type(4))) u32x4;   // plain vector (not HIP class)
typedef int      __attribute__((ext_vector_type(8))) int32x8;
typedef int      __attribute__((ext_vector_type(4))) int32x4;

// LDS (addrspace 3) pointer types so laundered pointers still select ds_load_*.
typedef __attribute__((address_space(3))) const bf16  lds_cbf;
typedef __attribute__((address_space(3))) const u32x4 lds_cu4;

union V16 { v16bf v; u32x4 u[2]; };

#if defined(__has_builtin)
#if __has_builtin(__builtin_amdgcn_tensor_load_to_lds) && \
    __has_builtin(__builtin_amdgcn_s_wait_tensorcnt)
#define USE_TDM 1
#endif
#endif
#ifndef USE_TDM
#define USE_TDM 0
#endif

__device__ __forceinline__ unsigned short bfu(float f) {
  unsigned u = __builtin_bit_cast(unsigned, f);
  u += 0x7FFFu + ((u >> 16) & 1u);           // round-to-nearest-even
  return (unsigned short)(u >> 16);
}
__device__ __forceinline__ bf16 f2bf(float f) {
  unsigned short h = bfu(f);
  return __builtin_bit_cast(bf16, h);
}
__device__ __forceinline__ v8f wmma_bf16(v16bf a, v16bf b, v8f c) {
  // (neg_a, A, neg_b, B, c_mod, C, reuse_a, reuse_b)
  return __builtin_amdgcn_wmma_f32_16x16x32_bf16(false, a, false, b, (short)0, c,
                                                 false, false);
}

// The TDM writes LDS behind the compiler's back (the builtin only carries a
// descriptor), so loads from the staging buffers would otherwise be folded to
// undef ("never stored"). Launder the LDS pointer through an opaque asm so the
// fragment ds_load_b128s are kept and ordered after the barrier.
__device__ __forceinline__ lds_cbf* opaque_lds(const bf16* p) {
  lds_cbf* q = (lds_cbf*)p;
  asm volatile("" : "+v"(q));
  return q;
}
__device__ __forceinline__ u32x4 lds_read16(lds_cbf* p) {
  return *(lds_cu4*)p;
}

// ---------------------------------------------------------------------------
// Stage 1: fp32 -> bf16 converters. Weights are written K-major (pre-transposed
// B^T) so both GEMM tiles are plain row-major [rows][32] slabs for the TDM.
// ---------------------------------------------------------------------------
__global__ void k_cvt4(const float* __restrict__ src, bf16* __restrict__ dst, int n4) {
  int i = blockIdx.x * blockDim.x + threadIdx.x;
  if (i >= n4) return;
  float4 f = ((const float4*)src)[i];
  ushort4 o;
  o.x = bfu(f.x); o.y = bfu(f.y); o.z = bfu(f.z); o.w = bfu(f.w);
  ((ushort4*)dst)[i] = o;
}

// WqkvT [3072][2048] bf16, row c = output column (q|k|v), col d = input dim.
__global__ void k_build_wqkvT(const float* __restrict__ Wq, const float* __restrict__ Wk,
                              const float* __restrict__ Wv, bf16* __restrict__ Wb, int n) {
  int i = blockIdx.x * blockDim.x + threadIdx.x;
  if (i >= n) return;
  int d = i % D_, c = i / D_;
  float v;
  if (c < NQ_ * H_)             v = Wq[(size_t)d * (NQ_ * H_) + c];
  else if (c < (NQ_ + NK_) * H_) v = Wk[(size_t)d * (NK_ * H_) + (c - NQ_ * H_)];
  else                           v = Wv[(size_t)d * (NK_ * H_) + (c - (NQ_ + NK_) * H_)];
  Wb[i] = f2bf(v);
}

// WoT [D][NQ*H] bf16 = transpose of Wo [NQ*H, D].
__global__ void k_cvt_woT(const float* __restrict__ Wo, bf16* __restrict__ WoT, int n) {
  int i = blockIdx.x * blockDim.x + threadIdx.x;
  if (i >= n) return;
  int nh = i % (NQ_ * H_), d = i / (NQ_ * H_);
  WoT[i] = f2bf(Wo[(size_t)nh * D_ + d]);
}

// ---------------------------------------------------------------------------
// Stage 2/5: bf16 WMMA GEMM  C[M,N](f32) = A[M,K] * Bt[N,K]^T   (bf16 inputs)
// Block tile 128x128, K-step 32, 256 threads = 8 waves.
// TDM path: wave 0 DMAs both 128x32 slabs into LDS (pad_enable gives the
// 40-element padded row pitch), double-buffered, one barrier per K-step,
// synced with s_wait_tensorcnt. Fallback: vectorized global->LDS staging.
// ---------------------------------------------------------------------------
#define LDSP 40
#define TILEB (128 * LDSP)                       // bf16 elements per tile slab
#define TILE_BYTES ((unsigned)(TILEB * 2))       // 10240 B

#if USE_TDM
__device__ __forceinline__ void tdm_load_tile(const bf16* gptr, unsigned lds_off,
                                              int kstride) {
  // 2D tile: tile_dim0 = 32 elems (2B), tile_dim1 = 128 rows, row stride = K.
  // LDS padding: +4 DWORDs (8 bf16) after every 16 DWORDs (32 bf16) -> LDSP 40.
  unsigned long long ga = (unsigned long long)gptr;
  union { unsigned d[4]; u32x4 v; } g0;
  g0.d[0] = 1u;                                              // count=1, user mode
  g0.d[1] = lds_off;                                         // lds_addr (bytes)
  g0.d[2] = (unsigned)ga;                                    // global_addr[31:0]
  g0.d[3] = (unsigned)((ga >> 32) & 0x01FFFFFFu) | (2u << 30);  // [56:32] | type=2
  const unsigned td0 = 0x40000000u, td1 = 0x40000000u;       // huge: no OOB clip
  union { int d[8]; int32x8 v; } g1;
  g1.d[0] = (int)((1u << 16)      // data_size = 2 bytes
                | (1u << 20)      // pad_enable
                | (3u << 22)      // pad_interval: every 16 DWORDs
                | (3u << 25));    // pad_amount: 4 DWORDs
  g1.d[1] = (int)((td0 & 0xFFFFu) << 16);                    // abar=0 | td0[15:0]
  g1.d[2] = (int)((td0 >> 16) | ((td1 & 0xFFFFu) << 16));    // td0[31:16] | td1[15:0]
  g1.d[3] = (int)((td1 >> 16) | (32u << 16));                // td1[31:16] | tile_dim0=32
  g1.d[4] = 128;                                             // tile_dim1=128, tile_dim2=0
  g1.d[5] = kstride;                                         // dim0 stride (elems) lo
  g1.d[6] = 0;                                               // stride hi | dim1 stride lo
  g1.d[7] = 0;
  int32x4 z4 = {0, 0, 0, 0};                                 // groups 2/3: 2-D tile
  int32x8 z8 = {0, 0, 0, 0, 0, 0, 0, 0};
  __builtin_amdgcn_tensor_load_to_lds(g0.v, g1.v, z4, z4, z8, 0);
}
#endif

__global__ __launch_bounds__(256) void k_gemm_bf16(const bf16* __restrict__ A,
                                                   const bf16* __restrict__ Bt,
                                                   float* __restrict__ C,
                                                   int K, int N) {
#if USE_TDM
  __shared__ __align__(16) bf16 smem[2][2][TILEB];   // [buf][A|B][slab]
#else
  __shared__ __align__(16) bf16 smem[1][2][TILEB];
#endif
  const int tid  = threadIdx.x;
  const int lane = tid & 31;
  const int wid  = tid >> 5;
  const int m0 = blockIdx.y * 128;
  const int n0 = blockIdx.x * 128;
  const int wm = (wid & 3) * 32;   // wave M offset in tile
  const int wn = (wid >> 2) * 64;  // wave N offset in tile
  const int lr = lane & 15;
  const int hi = lane >> 4;

  const bf16* Arow = A + (size_t)m0 * K;
  const bf16* Brow = Bt + (size_t)n0 * K;

  v8f acc[2][4];
#pragma unroll
  for (int fm = 0; fm < 2; ++fm)
#pragma unroll
    for (int fn = 0; fn < 4; ++fn)
#pragma unroll
      for (int i = 0; i < 8; ++i) acc[fm][fn][i] = 0.0f;

#if USE_TDM
  if (wid == 0) {                       // prime buffer 0
    tdm_load_tile(Arow, 0u, K);
    tdm_load_tile(Brow, TILE_BYTES, K);
  }
  int buf = 0;
#endif

  for (int k0 = 0; k0 < K; k0 += 32) {
#if USE_TDM
    if (wid == 0) __builtin_amdgcn_s_wait_tensorcnt(0);  // current buf in LDS
    __syncthreads();                                     // publish / prev consumed
    if (wid == 0 && k0 + 32 < K) {                       // DMA next buf now
      unsigned nb = (unsigned)(buf ^ 1) * 2u * TILE_BYTES;
      tdm_load_tile(Arow + k0 + 32, nb, K);
      tdm_load_tile(Brow + k0 + 32, nb + TILE_BYTES, K);
    }
    lds_cbf* As = opaque_lds(&smem[buf][0][0]);
    lds_cbf* Bs = opaque_lds(&smem[buf][1][0]);
#else
    // fallback: vectorized staging, both slabs are row-major [128][32]
#pragma unroll
    for (int j = 0; j < 2; ++j) {
      int idx = tid * 2 + j;                 // 0..511
      int r = idx >> 2, cc = (idx & 3) * 8;
      u32x4 da = *(const u32x4*)(Arow + (size_t)r * K + k0 + cc);
      *(u32x4*)(&smem[0][0][0] + r * LDSP + cc) = da;
      u32x4 db = *(const u32x4*)(Brow + (size_t)r * K + k0 + cc);
      *(u32x4*)(&smem[0][1][0] + r * LDSP + cc) = db;
    }
    __syncthreads();
    lds_cbf* As = opaque_lds(&smem[0][0][0]);
    lds_cbf* Bs = opaque_lds(&smem[0][1][0]);
#endif

    // A frags: lane row = wm+fm*16+lr; elems 0..7 -> k=8*hi.., 8..15 -> 16+8*hi..
    V16 af[2];
#pragma unroll
    for (int fm = 0; fm < 2; ++fm) {
      lds_cbf* p = As + (wm + fm * 16 + lr) * LDSP + 8 * hi;
      af[fm].u[0] = lds_read16(p);
      af[fm].u[1] = lds_read16(p + 16);
    }
#pragma unroll
    for (int fn = 0; fn < 4; ++fn) {
      // B frag: lane col = wn+fn*16+lr; elems 0..7 -> k=8*hi.., 8..15 -> 16+8*hi..
      V16 bfr;
      lds_cbf* p = Bs + (wn + fn * 16 + lr) * LDSP + 8 * hi;
      bfr.u[0] = lds_read16(p);
      bfr.u[1] = lds_read16(p + 16);
#pragma unroll
      for (int fm = 0; fm < 2; ++fm)
        acc[fm][fn] = wmma_bf16(af[fm].v, bfr.v, acc[fm][fn]);
    }
#if USE_TDM
    buf ^= 1;
#else
    __syncthreads();
#endif
  }

  // C frag layout: elem i -> row = i + 8*hi, col = lr. Pointer-stepped stores.
#pragma unroll
  for (int fm = 0; fm < 2; ++fm)
#pragma unroll
    for (int fn = 0; fn < 4; ++fn) {
      float* cp = C + (size_t)(m0 + wm + fm * 16 + 8 * hi) * N
                    + (n0 + wn + fn * 16 + lr);
#pragma unroll
      for (int i = 0; i < 8; ++i) { *cp = acc[fm][fn][i]; cp += N; }
    }
}

// ---------------------------------------------------------------------------
// Stage 3: RoPE epilogues reading fp32 QKV [BT, 3072]
//   Qb [B,NQ,T,H], Kb [B,NK,T,H], Vt [B,NK,H,T]  (V transposed for P@V frags)
// ---------------------------------------------------------------------------
__global__ void k_rope_q(const float* __restrict__ QKV, const int* __restrict__ pos,
                         bf16* __restrict__ Qb, int n) {
  int idx = blockIdx.x * blockDim.x + threadIdx.x;
  if (idx >= n) return;
  int i = idx & 63;
  int h0 = (idx >> 6) % NQ_;
  int bt = idx / (NQ_ * 64);
  int t = bt % T_, b = bt / T_;
  float p = (float)pos[bt];
  float inv = __powf(10000.0f, -(float)i * (1.0f / 64.0f));
  float s, c;
  __sincosf(p * inv, &s, &c);
  const float* row = QKV + (size_t)bt * CQKV_ + h0 * H_;
  float x1 = row[i], x2 = row[i + 64];
  size_t o = (((size_t)b * NQ_ + h0) * T_ + t) * H_;
  Qb[o + i]      = f2bf(x1 * c - x2 * s);
  Qb[o + i + 64] = f2bf(x2 * c + x1 * s);
}

__global__ void k_rope_k(const float* __restrict__ QKV, const int* __restrict__ pos,
                         bf16* __restrict__ Kb, int n) {
  int idx = blockIdx.x * blockDim.x + threadIdx.x;
  if (idx >= n) return;
  int i = idx & 63;
  int kh = (idx >> 6) % NK_;
  int bt = idx / (NK_ * 64);
  int t = bt % T_, b = bt / T_;
  float p = (float)pos[bt];
  float inv = __powf(10000.0f, -(float)i * (1.0f / 64.0f));
  float s, c;
  __sincosf(p * inv, &s, &c);
  const float* row = QKV + (size_t)bt * CQKV_ + NQ_ * H_ + kh * H_;
  float x1 = row[i], x2 = row[i + 64];
  size_t o = (((size_t)b * NK_ + kh) * T_ + t) * H_;
  Kb[o + i]      = f2bf(x1 * c - x2 * s);
  Kb[o + i + 64] = f2bf(x2 * c + x1 * s);
}

__global__ void k_cvt_v(const float* __restrict__ QKV, bf16* __restrict__ Vt, int n) {
  int idx = blockIdx.x * blockDim.x + threadIdx.x;
  if (idx >= n) return;
  int h = idx & (H_ - 1);
  int kh = (idx >> 7) % NK_;
  int bt = idx / (NK_ * H_);
  int t = bt % T_, b = bt / T_;
  float v = QKV[(size_t)bt * CQKV_ + (NQ_ + NK_) * H_ + kh * H_ + h];
  Vt[(((size_t)b * NK_ + kh) * H_ + h) * T_ + t] = f2bf(v);
}

// ---------------------------------------------------------------------------
// Stage 4: causal GQA flash attention (scale = 1.0, as in the reference).
// One wave per (b, qh, 16-row q tile); 8 independent waves per block.
// Transposed-logit trick: S^T = K_tile * Q^T, so the f32 C-fragment of S^T
// has exactly the bf16 A-fragment layout needed for P @ V (no LDS transpose).
// ---------------------------------------------------------------------------
#define NEG_BIG (-3.0e38f)

__global__ __launch_bounds__(256) void k_attn(const bf16* __restrict__ Qb,
                                              const bf16* __restrict__ Kb,
                                              const bf16* __restrict__ Vt,
                                              bf16* __restrict__ Ob) {
  const int lane = threadIdx.x & 31;
  const int wid  = threadIdx.x >> 5;
  const int MT   = T_ / 16;
  int gid = blockIdx.x * 8 + wid;
  int mt  = gid % MT;
  int qh  = (gid / MT) % NQ_;
  int b   = gid / (MT * NQ_);
  int kh  = qh >> 2;  // G = NQ/NK = 4
  const int lr = lane & 15, hi = lane >> 4;
  const int m0 = mt * 16;
  const int trow = m0 + lr;  // this lane's query position (logit column)

  // Q B-fragments (Q^T chunks): lane col = q row m0+lr, elems 0..7 -> h = 32c+8*hi.., 8..15 -> 32c+16+8*hi..
  const bf16* qbase = Qb + (((size_t)b * NQ_ + qh) * T_ + m0 + lr) * H_;
  V16 qf[4];
#pragma unroll
  for (int c = 0; c < 4; ++c) {
    const bf16* p = qbase + 32 * c + 8 * hi;
    qf[c].u[0] = *(const u32x4*)p;
    qf[c].u[1] = *(const u32x4*)(p + 16);
  }

  v8f oacc[8];
#pragma unroll
  for (int nc = 0; nc < 8; ++nc)
#pragma unroll
    for (int i = 0; i < 8; ++i) oacc[nc][i] = 0.0f;

  float mrow = NEG_BIG, lrow = 0.0f;
  const bf16* kbase = Kb + ((size_t)b * NK_ + kh) * T_ * H_;
  const bf16* vbase = Vt + ((size_t)b * NK_ + kh) * H_ * T_;

  for (int s0 = 0; s0 <= m0; s0 += 32) {
    // ---- logits^T: two 16(s) x 16(m) tiles, K-dim = H = 4 chunks of 32
    v8f c0, c1;
#pragma unroll
    for (int i = 0; i < 8; ++i) { c0[i] = 0.0f; c1[i] = 0.0f; }
#pragma unroll
    for (int c = 0; c < 4; ++c) {
      V16 kf;
      const bf16* p = kbase + (size_t)(s0 + lr) * H_ + 32 * c + 8 * hi;
      kf.u[0] = *(const u32x4*)p;
      kf.u[1] = *(const u32x4*)(p + 16);
      c0 = wmma_bf16(kf.v, qf[c].v, c0);
      const bf16* p1 = p + 16 * H_;
      kf.u[0] = *(const u32x4*)p1;
      kf.u[1] = *(const u32x4*)(p1 + 16);
      c1 = wmma_bf16(kf.v, qf[c].v, c1);
    }
    if (s0 + 32 <= m0)  // global_prefetch_b8 of next K tile
      __builtin_prefetch(kbase + (size_t)(s0 + 32 + lr) * H_, 0, 1);

    // ---- causal mask + online softmax (row = lane's column trow)
    float tmax = NEG_BIG;
#pragma unroll
    for (int i = 0; i < 8; ++i) {
      int s = s0 + i + 8 * hi;          // c0's key position; c1's is s+16
      if (s > trow)      c0[i] = NEG_BIG;
      if (s + 16 > trow) c1[i] = NEG_BIG;
      tmax = fmaxf(tmax, fmaxf(c0[i], c1[i]));
    }
    tmax = fmaxf(tmax, __shfl_xor(tmax, 16, 32));   // wave32 half-reduction
    float mnew  = fmaxf(mrow, tmax);
    float alpha = __expf(mrow - mnew);
    mrow = mnew;

    V16 pa;                                         // P as bf16 A-fragment
    float psum = 0.0f;
#pragma unroll
    for (int i = 0; i < 8; ++i) {
      float p0 = __expf(c0[i] - mnew);
      float p1 = __expf(c1[i] - mnew);
      psum += p0 + p1;
      pa.v[i]     = f2bf(p0);
      pa.v[i + 8] = f2bf(p1);
    }
    psum += __shfl_xor(psum, 16, 32);
    lrow = lrow * alpha + psum;

    // ---- rescale O (row of elem i is i+8*hi; stats live in lane == row)
    float al[8];
#pragma unroll
    for (int i = 0; i < 8; ++i) al[i] = __shfl(alpha, i + 8 * hi, 32);
#pragma unroll
    for (int nc = 0; nc < 8; ++nc)
#pragma unroll
      for (int i = 0; i < 8; ++i) oacc[nc][i] *= al[i];

    // ---- O += P @ V  (V^T rows are contiguous in s: perfect B-fragments)
#pragma unroll
    for (int nc = 0; nc < 8; ++nc) {
      V16 vf;
      const bf16* p = vbase + (size_t)(16 * nc + lr) * T_ + s0 + 8 * hi;
      vf.u[0] = *(const u32x4*)p;
      vf.u[1] = *(const u32x4*)(p + 16);
      oacc[nc] = wmma_bf16(pa.v, vf.v, oacc[nc]);
    }
  }

  // ---- finalize: O /= l, write bf16 attn [B,T,NQ,H] (= [BT, 2048] row-major)
  float rl[8];
#pragma unroll
  for (int i = 0; i < 8; ++i) {
    float v = __shfl(lrow, i + 8 * hi, 32);
    rl[i] = 1.0f / v;
  }
  bf16* obase = Ob + ((size_t)(b * T_ + m0) * NQ_ + qh) * H_;
#pragma unroll
  for (int nc = 0; nc < 8; ++nc) {
    int h = 16 * nc + lr;
#pragma unroll
    for (int i = 0; i < 8; ++i) {
      int m = i + 8 * hi;
      obase[(size_t)m * (NQ_ * H_) + h] = f2bf(oacc[nc][i] * rl[i]);
    }
  }
}

// ---------------------------------------------------------------------------
// Launcher. Workspace carve-up (~130 MB):
//   Xb 16.8M | WqkvT 12.6M | QKV(f32) 50.3M | Qb 16.8M | Kb 4.2M+pad
//   | Vt 4.2M+pad | Attnb 16.8M | WoT 8.4M
// ---------------------------------------------------------------------------
extern "C" void kernel_launch(void* const* d_in, const int* in_sizes, int n_in,
                              void* d_out, int out_size, void* d_ws, size_t ws_size,
                              hipStream_t stream) {
  const float* X    = (const float*)d_in[0];
  const int*   qpos = (const int*)d_in[1];
  const float* Wq   = (const float*)d_in[2];
  const float* Wk   = (const float*)d_in[3];
  const float* Wv   = (const float*)d_in[4];
  const float* Wo   = (const float*)d_in[5];
  float* out = (float*)d_out;

  char* p = (char*)d_ws;
  auto carve = [&](size_t bytes) {
    char* r = p;
    p += (bytes + 255) & ~(size_t)255;
    return r;
  };
  bf16*  Xb    = (bf16*)carve((size_t)BT_ * D_ * 2);
  bf16*  WqkvT = (bf16*)carve((size_t)CQKV_ * D_ * 2);
  float* QKV   = (float*)carve((size_t)BT_ * CQKV_ * 4);
  bf16*  Qb    = (bf16*)carve((size_t)B_ * NQ_ * T_ * H_ * 2);
  bf16*  Kb    = (bf16*)carve((size_t)B_ * NK_ * T_ * H_ * 2 + 8192);  // +tail pad
  bf16*  Vt    = (bf16*)carve((size_t)B_ * NK_ * H_ * T_ * 2 + 8192);  // +tail pad
  bf16*  Attnb = (bf16*)carve((size_t)BT_ * (NQ_ * H_) * 2);
  bf16*  WoT   = (bf16*)carve((size_t)D_ * (NQ_ * H_) * 2);

  // 1) conversions (weights transposed to [N][K] for the GEMM B-side)
  {
    int n4 = BT_ * D_ / 4;
    k_cvt4<<<n4 / 256, 256, 0, stream>>>(X, Xb, n4);
  }
  {
    int n = CQKV_ * D_;
    k_build_wqkvT<<<(n + 255) / 256, 256, 0, stream>>>(Wq, Wk, Wv, WqkvT, n);
  }
  {
    int n = D_ * (NQ_ * H_);
    k_cvt_woT<<<(n + 255) / 256, 256, 0, stream>>>(Wo, WoT, n);
  }

  // 2) fused QKV projection: [4096,2048] x [2048,3072] -> f32
  k_gemm_bf16<<<dim3(CQKV_ / 128, BT_ / 128), 256, 0, stream>>>(Xb, WqkvT, QKV, D_, CQKV_);

  // 3) RoPE + layout epilogues
  {
    int n = BT_ * NQ_ * 64;
    k_rope_q<<<n / 256, 256, 0, stream>>>(QKV, qpos, Qb, n);
  }
  {
    int n = BT_ * NK_ * 64;
    k_rope_k<<<n / 256, 256, 0, stream>>>(QKV, qpos, Kb, n);
  }
  {
    int n = BT_ * NK_ * H_;
    k_cvt_v<<<n / 256, 256, 0, stream>>>(QKV, Vt, n);
  }

  // 4) flash attention: B*NQ*(T/16) = 4096 waves -> 512 blocks of 8 waves
  k_attn<<<(B_ * NQ_ * (T_ / 16)) / 8, 256, 0, stream>>>(Qb, Kb, Vt, Attnb);

  // 5) output projection: [4096,2048] x [2048,2048] -> d_out (f32)
  k_gemm_bf16<<<dim3(D_ / 128, BT_ / 128), 256, 0, stream>>>(Attnb, WoT, out, NQ_ * H_, D_);
}
